// LSTMNet_20444044329524
// MI455X (gfx1250) — matrix operand order repaired and checked by reading, not compiled, not run on hardware
//
#include <hip/hip_runtime.h>
#include <hip/hip_bf16.h>
#include <stdint.h>

#define NWG     16
#define SEQ_LEN 2048
#define BATCH_N 64
#define HID     256

#if defined(__has_builtin)
#if __has_builtin(__builtin_amdgcn_global_load_async_to_lds_b128) && \
    __has_builtin(__builtin_amdgcn_s_wait_asynccnt)
#define HAVE_ASYNC_LDS 1
#endif
#endif

typedef __attribute__((ext_vector_type(16))) __bf16 bf16x16;
typedef __attribute__((ext_vector_type(8)))  float  f32x8;
typedef int v4i_ __attribute__((vector_size(16)));   // matches builtin param type
typedef __attribute__((address_space(1))) v4i_* gv4i_p;   // global int4*
typedef __attribute__((address_space(3))) v4i_* lv4i_p;   // LDS int4*

union Frag16 { bf16x16 v; uint32_t u[8]; uint4 q[2]; };

__device__ __forceinline__ uint16_t f32_to_bf16(float f) {
  union { float f; uint32_t u; } x; x.f = f;
  uint32_t r = x.u + 0x7FFFu + ((x.u >> 16) & 1u);   // round-to-nearest-even
  return (uint16_t)(r >> 16);
}
__device__ __forceinline__ float bf16_to_f32(uint16_t h) {
  union { uint32_t u; float f; } x; x.u = ((uint32_t)h) << 16;
  return x.f;
}
__device__ __forceinline__ float sigmoidf_(float x) {
  return 1.0f / (1.0f + __expf(-x));
}

// ---------------------------------------------------------------- prep kernels

__global__ void zero_counters_kernel(uint32_t* c) {
  if (threadIdx.x < 128) c[threadIdx.x] = 0u;
}

__global__ void convert_f32_bf16_kernel(const float* __restrict__ src,
                                        uint16_t* __restrict__ dst, int n) {
  int i = blockIdx.x * blockDim.x + threadIdx.x;
  if (i < n) dst[i] = f32_to_bf16(src[i]);
}

// Pack W [1024 x Dk] (gate-major rows i,f,g,o) into per-WG WMMA B fragments:
// dst dword index = (((w*4 + gate)*KT + kt)*32 + lane)*8 + d
// B fragment element e on lane L: N = L&15, K = (e&7) + 8*(L>=16) + 16*(e>=8); dword d = e/2.
__global__ void pack_w_kernel(const float* __restrict__ W,
                              uint32_t* __restrict__ dst, int Dk) {
  int idx = blockIdx.x * blockDim.x + threadIdx.x;
  int KT = Dk >> 5;
  int total = 512 * Dk;                 // 16*4*KT*32*8 dwords
  if (idx >= total) return;
  int d    = idx & 7;
  int lane = (idx >> 3) & 31;
  int rest = idx >> 8;
  int kt = rest % KT; rest /= KT;
  int gt = rest & 3;
  int w  = rest >> 2;
  int col = lane & 15;
  int r = gt * 256 + w * 16 + col;      // gate row in [0,1024)
  int K = kt * 32 + 2 * (d & 3) + ((d >> 2) << 4) + ((lane >> 4) << 3);
  uint32_t lo = f32_to_bf16(W[r * Dk + K]);
  uint32_t hi = f32_to_bf16(W[r * Dk + K + 1]);
  dst[idx] = lo | (hi << 16);
}

// ---------------------------------------------------------------- LSTM scan
// 16 persistent workgroups; WG w owns h-columns [w*16, w*16+16) i.e. gate rows
// {g*256 + w*16 + j}. Per step:
//   1) issue async global->LDS copy of h_{t-1}, permuted into A-fragment order
//   2) x_t @ WihS^T with register-resident Wih fragments (overlaps the DMA)
//   3) wait asynccnt + barrier, h_{t-1} @ WhhS^T from LDS fragments
//   4) cell update (c in registers), bf16 h slice to double-buffered global h
//   5) device-scope arrive/spin barrier
template<int DK>
__global__ __launch_bounds__(256, 1)
void lstm_scan_kernel(const uint16_t* __restrict__ xseq,   // [SEQ][64][DK] bf16
                      const uint32_t* __restrict__ wih_pk, // packed fragments
                      const uint32_t* __restrict__ whh_pk, // packed fragments
                      const float*    __restrict__ bias,   // [1024] f32
                      uint16_t*       __restrict__ hbuf,   // [2][64][256] bf16
                      uint16_t*       __restrict__ seq_out,// [SEQ][64][256] bf16 (tanh h)
                      uint32_t*       __restrict__ sync_cnt) {
  constexpr int KT_IN = DK / 32;
  const int wg   = blockIdx.x;          // 0..15
  const int tid  = threadIdx.x;
  const int lane = tid & 31;            // wave32
  const int wv   = tid >> 5;            // 8 waves
  const int mt   = wv & 3;              // batch tile (4 x 16 rows)
  const int gt0  = wv >> 2;             // this wave: gates {gt0, gt0+2}
  const int gt1  = gt0 + 2;
  const int b_row = mt * 16 + (lane & 15);   // A-fragment row (batch index)
  const int koff  = (lane >= 16) ? 8 : 0;    // A layout K offset per lane half

  __shared__ __align__(16) uint32_t whh_lds[4 * 8 * 32 * 8]; // 32 KB packed Whh slice
  __shared__ __align__(16) uint32_t h_lds[4 * 8 * 32 * 8];   // 32 KB h in A-frag order
  __shared__ float gates_lds[4 * 64 * 16];                   // 16 KB

  // one-time: stage this WG's packed Whh slice into LDS
  const uint32_t* whh_src = whh_pk + (size_t)wg * (4 * 8 * 32 * 8);
  for (int i = tid; i < 4 * 8 * 32 * 8; i += 256) whh_lds[i] = whh_src[i];
  __syncthreads();

  // one-time: hoist this wave's Wih B fragments into registers (loop invariant)
  Frag16 wihB0[KT_IN], wihB1[KT_IN];
  {
    const uint4* wih_pk4 = (const uint4*)wih_pk;
    #pragma unroll
    for (int kt = 0; kt < KT_IN; ++kt) {
      const uint4* p0 = wih_pk4 + ((((wg * 4 + gt0) * KT_IN + kt) * 32 + lane) << 1);
      const uint4* p1 = wih_pk4 + ((((wg * 4 + gt1) * KT_IN + kt) * 32 + lane) << 1);
      wihB0[kt].q[0] = p0[0]; wihB0[kt].q[1] = p0[1];
      wihB1[kt].q[0] = p1[0]; wihB1[kt].q[1] = p1[1];
    }
  }

  float c_reg[4] = {0.f, 0.f, 0.f, 0.f};   // cell state, fixed (b,j) per thread

  for (int t = 0; t < SEQ_LEN; ++t) {
    // -------- 1) stage h_{t-1} into LDS in A-fragment order (overlaps step 2)
    // chunk c (0..2047): one b128 = 4 dwords. dest ((cmt*8+ckt)*32+clane)*8 + half*4
    // src: 8 consecutive bf16 at row b=cmt*16+(clane&15), K=ckt*32+16*half+8*(clane>=16)
    if (t > 0) {
      const uint32_t* hp = (const uint32_t*)(hbuf + (size_t)(t & 1) * (BATCH_N * HID));
      #pragma unroll
      for (int i = 0; i < 8; ++i) {
        int c     = i * 256 + tid;
        int half  = c & 1;
        int clane = (c >> 1) & 31;
        int ckt   = (c >> 6) & 7;
        int cmt   = (c >> 9) & 3;
        int brow  = cmt * 16 + (clane & 15);
        int Kb    = ckt * 32 + (half << 4) + ((clane >= 16) ? 8 : 0);
        int src_dw = (brow * HID + Kb) >> 1;
        int dst_dw = (((cmt * 8 + ckt) * 32 + clane) << 3) + (half << 2);
#ifdef HAVE_ASYNC_LDS
        __builtin_amdgcn_global_load_async_to_lds_b128(
            (gv4i_p)(hp + src_dw), (lv4i_p)(h_lds + dst_dw), 0, 0);
#else
        *(uint4*)(h_lds + dst_dw) = *(const uint4*)(hp + src_dw);
#endif
      }
    }

    f32x8 acc0 = {}; f32x8 acc1 = {};

    // -------- 2) input projection x_t @ WihS^T (Wih fragments in registers)
    {
      const uint32_t* xt = (const uint32_t*)(xseq + (size_t)t * (BATCH_N * DK));
      Frag16 xa[KT_IN];
      #pragma unroll
      for (int kt = 0; kt < KT_IN; ++kt) {        // issue all loads first
        #pragma unroll
        for (int d = 0; d < 8; ++d) {
          int K = kt * 32 + 2 * d + ((d >= 4) ? 8 : 0) + koff; // 16x32 bf16 A layout
          xa[kt].u[d] = xt[(b_row * DK + K) >> 1];
        }
      }
      #pragma unroll
      for (int kt = 0; kt < KT_IN; ++kt) {
        acc0 = __builtin_amdgcn_wmma_f32_16x16x32_bf16(false, xa[kt].v, false, wihB0[kt].v, (short)0, acc0, false, false);
        acc1 = __builtin_amdgcn_wmma_f32_16x16x32_bf16(false, xa[kt].v, false, wihB1[kt].v, (short)0, acc1, false, false);
      }
    }

    // prefetch next timestep's x rows while we finish this step
    if (t + 1 < SEQ_LEN)
      __builtin_prefetch(xseq + (size_t)(t + 1) * (BATCH_N * DK) + b_row * DK, 0, 1);

    // -------- 3) recurrent projection h_{t-1} @ WhhS^T from LDS (h0 == 0)
    if (t > 0) {
#ifdef HAVE_ASYNC_LDS
      __builtin_amdgcn_s_wait_asynccnt(0);
#endif
      __syncthreads();   // h_lds fully staged by all waves
      #pragma unroll
      for (int kt = 0; kt < 8; ++kt) {
        Frag16 a;
        const uint4* ap = (const uint4*)&h_lds[(((mt * 8 + kt) * 32 + lane) << 3)];
        a.q[0] = ap[0]; a.q[1] = ap[1];
        Frag16 b0, b1;
        const uint4* q0 = (const uint4*)&whh_lds[(((gt0 * 8 + kt) * 32 + lane) << 3)];
        const uint4* q1 = (const uint4*)&whh_lds[(((gt1 * 8 + kt) * 32 + lane) << 3)];
        b0.q[0] = q0[0]; b0.q[1] = q0[1];
        b1.q[0] = q1[0]; b1.q[1] = q1[1];
        acc0 = __builtin_amdgcn_wmma_f32_16x16x32_bf16(false, a.v, false, b0.v, (short)0, acc0, false, false);
        acc1 = __builtin_amdgcn_wmma_f32_16x16x32_bf16(false, a.v, false, b1.v, (short)0, acc1, false, false);
      }
    }

    // -------- scatter accumulators (16x16 f32 C layout) to LDS
    {
      const int rb  = (lane >= 16) ? 8 : 0;
      const int col = lane & 15;
      #pragma unroll
      for (int r = 0; r < 8; ++r) {
        int bb = mt * 16 + r + rb;
        gates_lds[(gt0 * 64 + bb) * 16 + col] = acc0[r];
        gates_lds[(gt1 * 64 + bb) * 16 + col] = acc1[r];
      }
    }
    __syncthreads();

    // -------- 4) LSTM cell update for this WG's 16 h-columns
    uint16_t* hb_new = hbuf + (size_t)((t + 1) & 1) * (BATCH_N * HID);
    uint16_t* so     = seq_out + (size_t)t * (BATCH_N * HID);
    #pragma unroll
    for (int qd = 0; qd < 4; ++qd) {
      int p  = qd * 256 + tid;
      int bb = p >> 4;
      int j  = p & 15;
      int gcol = wg * 16 + j;
      float gi = gates_lds[(0 * 64 + bb) * 16 + j] + bias[0 * 256 + gcol];
      float gf = gates_lds[(1 * 64 + bb) * 16 + j] + bias[1 * 256 + gcol];
      float gg = gates_lds[(2 * 64 + bb) * 16 + j] + bias[2 * 256 + gcol];
      float go = gates_lds[(3 * 64 + bb) * 16 + j] + bias[3 * 256 + gcol];
      gi = sigmoidf_(gi); gf = sigmoidf_(gf); go = sigmoidf_(go);
      gg = tanhf(gg);
      float c = gf * c_reg[qd] + gi * gg;
      c_reg[qd] = c;
      float h = go * tanhf(c);
      hb_new[bb * HID + gcol] = f32_to_bf16(h);         // recurrence carries h
      so[bb * HID + gcol]     = f32_to_bf16(tanhf(h));  // layer output = tanh(h)
    }

    // -------- 5) device-wide per-step barrier (arrive + spin, monotonic counter)
    __threadfence();
    __syncthreads();
    if (tid == 0) {
      __hip_atomic_fetch_add(sync_cnt, 1u, __ATOMIC_RELEASE, __HIP_MEMORY_SCOPE_AGENT);
      uint32_t need = (uint32_t)NWG * (uint32_t)(t + 1);
      while (__hip_atomic_load(sync_cnt, __ATOMIC_ACQUIRE, __HIP_MEMORY_SCOPE_AGENT) < need)
        __builtin_amdgcn_s_sleep(2);
    }
    __syncthreads();
  }
}

// ---------------------------------------------------------------- output head
__global__ __launch_bounds__(256)
void linear_sigmoid_kernel(const uint16_t* __restrict__ seq,  // [S*B][256] bf16
                           const float* __restrict__ Wlin,    // [10][256]
                           const float* __restrict__ blin,    // [10]
                           float* __restrict__ out) {         // [S*B][10]
  __shared__ float wl[10 * 256];
  for (int i = threadIdx.x; i < 10 * 256; i += 256) wl[i] = Wlin[i];
  __syncthreads();
  int row = blockIdx.x * 256 + threadIdx.x;   // < 131072
  const uint16_t* s = seq + (size_t)row * HID;
  float acc[10];
  #pragma unroll
  for (int c = 0; c < 10; ++c) acc[c] = blin[c];
  for (int k = 0; k < HID; ++k) {
    float v = bf16_to_f32(s[k]);
    #pragma unroll
    for (int c = 0; c < 10; ++c) acc[c] += v * wl[c * 256 + k];
  }
  float* o = out + (size_t)row * 10;
  #pragma unroll
  for (int c = 0; c < 10; ++c) o[c] = sigmoidf_(acc[c]);
}

// ---------------------------------------------------------------- launch
extern "C" void kernel_launch(void* const* d_in, const int* in_sizes, int n_in,
                              void* d_out, int out_size, void* d_ws, size_t ws_size,
                              hipStream_t stream) {
  (void)in_sizes; (void)n_in; (void)out_size; (void)ws_size;
  const float* x    = (const float*)d_in[0];
  const float* Wih1 = (const float*)d_in[1];
  const float* Whh1 = (const float*)d_in[2];
  const float* b1   = (const float*)d_in[3];
  const float* Wih2 = (const float*)d_in[4];
  const float* Whh2 = (const float*)d_in[5];
  const float* b2   = (const float*)d_in[6];
  const float* Wlin = (const float*)d_in[7];
  const float* blin = (const float*)d_in[8];
  float* out = (float*)d_out;

  uint8_t* ws = (uint8_t*)d_ws;
  size_t off = 0;
  auto walloc = [&](size_t bytes) -> void* {
    void* p = ws + off;
    off = (off + bytes + 255) & ~(size_t)255;
    return p;
  };
  uint16_t* xseq1 = (uint16_t*)walloc(2048ull * 64 * 128 * 2); //  32 MB bf16 x
  uint16_t* seq1  = (uint16_t*)walloc(2048ull * 64 * 256 * 2); //  64 MB tanh(h1)
  uint16_t* seq2  = (uint16_t*)walloc(2048ull * 64 * 256 * 2); //  64 MB tanh(h2)
  uint16_t* hbuf  = (uint16_t*)walloc(2ull * 64 * 256 * 2);    //  64 KB dbl-buffered h
  uint32_t* wih1p = (uint32_t*)walloc(512ull * 128 * 4);       // packed fragments
  uint32_t* whh1p = (uint32_t*)walloc(512ull * 256 * 4);
  uint32_t* wih2p = (uint32_t*)walloc(512ull * 256 * 4);
  uint32_t* whh2p = (uint32_t*)walloc(512ull * 256 * 4);
  uint32_t* cnts  = (uint32_t*)walloc(512);                    // sync counters

  zero_counters_kernel<<<1, 128, 0, stream>>>(cnts);

  {
    int n = 2048 * 64 * 128;
    convert_f32_bf16_kernel<<<(n + 255) / 256, 256, 0, stream>>>(x, xseq1, n);
  }
  pack_w_kernel<<<(512 * 128 + 255) / 256, 256, 0, stream>>>(Wih1, wih1p, 128);
  pack_w_kernel<<<(512 * 256 + 255) / 256, 256, 0, stream>>>(Whh1, whh1p, 256);
  pack_w_kernel<<<(512 * 256 + 255) / 256, 256, 0, stream>>>(Wih2, wih2p, 256);
  pack_w_kernel<<<(512 * 256 + 255) / 256, 256, 0, stream>>>(Whh2, whh2p, 256);

  lstm_scan_kernel<128><<<NWG, 256, 0, stream>>>(xseq1, wih1p, whh1p, b1, hbuf, seq1, cnts + 0);
  lstm_scan_kernel<256><<<NWG, 256, 0, stream>>>(seq1,  wih2p, whh2p, b2, hbuf, seq2, cnts + 64);

  linear_sigmoid_kernel<<<(2048 * 64) / 256, 256, 0, stream>>>(seq2, Wlin, blin, out);
}
